// MolGNN_37349035606232
// MI455X (gfx1250) — matrix operand order repaired and checked by reading, not compiled, not run on hardware
//
#include <hip/hip_runtime.h>
#include <hip/hip_bf16.h>
#include <math.h>

// ---------------------------------------------------------------------------
// Types for CDNA5 WMMA (wave32)
// ---------------------------------------------------------------------------
typedef __attribute__((ext_vector_type(16))) __bf16 bf16x16;
typedef __attribute__((ext_vector_type(8)))  float  f32x8;

union FragAB { bf16x16 v; uint4 q[2]; };

__device__ __forceinline__ float geluf(float x) {
    return 0.5f * x * (1.0f + erff(x * 0.70710678118654752f));
}

__device__ __forceinline__ void atomicMaxF(float* addr, float val) {
    int old = __float_as_int(*addr);
    while (__int_as_float(old) < val) {
        int assumed = old;
        old = atomicCAS((int*)addr, assumed, __float_as_int(val));
        if (old == assumed) break;
    }
}

// ---------------------------------------------------------------------------
// Weight convert + transpose: W[K][N] f32  ->  Wt[N][Kp] bf16 (zero-padded K)
// ---------------------------------------------------------------------------
__global__ void k_wconv(const float* __restrict__ W, __bf16* __restrict__ Wt,
                        int K, int Nn, int Kp) {
    long idx = (long)blockIdx.x * 256 + threadIdx.x;
    long total = (long)Nn * Kp;
    if (idx >= total) return;
    int n = (int)(idx / Kp), k = (int)(idx % Kp);
    float v = (k < K) ? W[(size_t)k * Nn + n] : 0.0f;
    Wt[(size_t)n * Kp + k] = (__bf16)v;
}

// ---------------------------------------------------------------------------
// Atom embedding gather -> Xcat bf16 [N, 448] (432 valid + zero pad)
// ---------------------------------------------------------------------------
__global__ void k_embed_atoms(const int* __restrict__ x,
                              const float* e0, const float* e1, const float* e2,
                              const float* e3, const float* e4, const float* e5,
                              const float* e6, const float* e7, const float* e8,
                              __bf16* __restrict__ Xcat, int N) {
    long idx = (long)blockIdx.x * 256 + threadIdx.x;
    long total = (long)N * 448;
    if (idx >= total) return;
    int i = (int)(idx / 448), c = (int)(idx % 448);
    float v = 0.0f;
    if (c < 432) {
        const float* embs[9] = {e0,e1,e2,e3,e4,e5,e6,e7,e8};
        int j = c / 48, d = c % 48;
        v = embs[j][(size_t)x[(size_t)i * 9 + j] * 48 + d];
    }
    Xcat[(size_t)i * 448 + c] = (__bf16)v;
}

// Edge embedding sum -> Ein bf16 [E, 64] (48 valid + zero pad)
__global__ void k_embed_edges(const int* __restrict__ ea,
                              const float* e0, const float* e1, const float* e2,
                              __bf16* __restrict__ Ein, int E) {
    long idx = (long)blockIdx.x * 256 + threadIdx.x;
    long total = (long)E * 64;
    if (idx >= total) return;
    int i = (int)(idx / 64), c = (int)(idx % 64);
    float v = 0.0f;
    if (c < 48) {
        v = e0[(size_t)ea[(size_t)i*3 + 0] * 48 + c]
          + e1[(size_t)ea[(size_t)i*3 + 1] * 48 + c]
          + e2[(size_t)ea[(size_t)i*3 + 2] * 48 + c];
    }
    Ein[(size_t)i * 64 + c] = (__bf16)v;
}

// ---------------------------------------------------------------------------
// Tile loaders.
//   As tile: [128 rows][32 K] bf16, LDS row stride 40 (16B aligned, no bank dup)
//   Bs tile: [256 rows][32 K] bf16, LDS row stride 40
// f32 A: manual load + convert. bf16 A and Wt: async DMA global->LDS
// (`global_load_async_to_lds_b128`), tracked by ASYNCcnt.
// ---------------------------------------------------------------------------
__device__ __forceinline__ void load_tileA(const float* __restrict__ A, int lda,
                                           int m0, int k0, int M, int tid,
                                           __bf16* As) {
    #pragma unroll
    for (int u = tid; u < 1024; u += 256) {      // 128 rows * 8 float4
        int row = u >> 3, c4 = (u & 7) * 4;
        int grow = m0 + row; if (grow >= M) grow = M - 1;   // clamp: row never stored
        float4 f = *reinterpret_cast<const float4*>(A + (size_t)grow * lda + k0 + c4);
        __bf16* d = As + row * 40 + c4;
        d[0] = (__bf16)f.x; d[1] = (__bf16)f.y; d[2] = (__bf16)f.z; d[3] = (__bf16)f.w;
    }
}

__device__ __forceinline__ void load_tileA(const __bf16* __restrict__ A, int lda,
                                           int m0, int k0, int M, int tid,
                                           __bf16* As) {
    #pragma unroll
    for (int u = tid; u < 512; u += 256) {       // 128 rows * 4 x 16B
        int row = u >> 2, c4 = (u & 3) * 8;
        int grow = m0 + row; if (grow >= M) grow = M - 1;
        const __bf16* src = A + (size_t)grow * lda + k0 + c4;
        unsigned ldsoff = (unsigned)(size_t)(As + row * 40 + c4);
        asm volatile("global_load_async_to_lds_b128 %0, %1, off"
                     :: "v"(ldsoff), "v"(src) : "memory");
    }
}

__device__ __forceinline__ void load_tileB(const __bf16* __restrict__ Wt, int K,
                                           int n0, int k0, int tid, __bf16* Bs) {
    #pragma unroll
    for (int u = tid; u < 1024; u += 256) {      // 256 rows * 4 x 16B (Nn % 256 == 0)
        int nrow = u >> 2, c4 = (u & 3) * 8;
        const __bf16* src = Wt + (size_t)(n0 + nrow) * K + k0 + c4;
        unsigned ldsoff = (unsigned)(size_t)(Bs + nrow * 40 + c4);
        asm volatile("global_load_async_to_lds_b128 %0, %1, off"
                     :: "v"(ldsoff), "v"(src) : "memory");
    }
}

// ---------------------------------------------------------------------------
// Tiled GEMM:  C[M,N] = epi(A[M,K] * Wt[N,K]^T + bias)
// block 256 thr = 8 waves, tile 128(M) x 256(N), K-step 32, double-buffered LDS,
// bf16 WMMA with f32 accumulation. Waves 4(M) x 2(N), 2x8 16x16 accs per wave.
// ---------------------------------------------------------------------------
template<bool GELU, typename AT, typename OT>
__global__ __launch_bounds__(256)
void k_gemm(const AT* __restrict__ A, int lda,
            const __bf16* __restrict__ Wt, const float* __restrict__ bias,
            OT* __restrict__ C, int ldc, int M, int Nn, int K) {
    __shared__ __bf16 As[2][128 * 40];
    __shared__ __bf16 Bs[2][256 * 40];

    const int tid  = threadIdx.x;
    const int lane = tid & 31;
    const int wave = tid >> 5;
    const int wm = (wave & 3) * 32;        // wave M offset in block tile
    const int wn = (wave >> 2) * 128;      // wave N offset in block tile
    const int hl  = lane >> 4;             // lane half (0/1)
    const int l16 = lane & 15;

    const int m0 = blockIdx.y * 128;
    const int n0 = blockIdx.x * 256;
    const int KT = K >> 5;                 // K is a multiple of 32 (padded)

    f32x8 acc[2][8];
    #pragma unroll
    for (int i = 0; i < 2; i++)
        #pragma unroll
        for (int j = 0; j < 8; j++)
            #pragma unroll
            for (int r = 0; r < 8; r++) acc[i][j][r] = 0.0f;

    // prime pipeline: tile 0 -> buffer 0
    load_tileA(A, lda, m0, 0, M, tid, As[0]);
    load_tileB(Wt, K, n0, 0, tid, Bs[0]);

    for (int kt = 0; kt < KT; kt++) {
        const int cur = kt & 1;
        asm volatile("s_wait_asynccnt 0x0" ::: "memory");  // async DMA of tile kt done
        __syncthreads();                                   // (implies s_wait_dscnt 0)

        if (kt + 1 < KT) {                                 // overlap: issue tile kt+1
            load_tileA(A, lda, m0, (kt + 1) * 32, M, tid, As[cur ^ 1]);
            load_tileB(Wt, K, n0, (kt + 1) * 32, tid, Bs[cur ^ 1]);
        }

        FragAB a[2];
        #pragma unroll
        for (int s = 0; s < 2; s++) {
            // ISA 16-bit A layout: lanes0-15 K0-7 & K16-23, lanes16-31 K8-15 & K24-31
            const __bf16* p = &As[cur][(wm + s * 16 + l16) * 40];
            a[s].q[0] = *reinterpret_cast<const uint4*>(p + hl * 8);
            a[s].q[1] = *reinterpret_cast<const uint4*>(p + 16 + hl * 8);
        }
        #pragma unroll
        for (int j = 0; j < 8; j++) {
            // 16-bit B layout (matches A): lanes0-15 K0-7 & K16-23, lanes16-31 K8-15 & K24-31 (N = lane%16)
            FragAB b;
            const __bf16* p = &Bs[cur][(wn + j * 16 + l16) * 40 + hl * 8];
            b.q[0] = *reinterpret_cast<const uint4*>(p);
            b.q[1] = *reinterpret_cast<const uint4*>(p + 16);
            acc[0][j] = __builtin_amdgcn_wmma_f32_16x16x32_bf16(
                false, a[0].v, false, b.v, (short)0, acc[0][j], false, false);
            acc[1][j] = __builtin_amdgcn_wmma_f32_16x16x32_bf16(
                false, a[1].v, false, b.v, (short)0, acc[1][j], false, false);
        }
    }

    // Epilogue. C/D layout: lane half selects M base (0/8), VGPR r -> M=half*8+r, N=lane%16
    #pragma unroll
    for (int i = 0; i < 2; i++) {
        #pragma unroll
        for (int j = 0; j < 8; j++) {
            const int nCol = n0 + wn + j * 16 + l16;
            const float bv = bias[nCol];
            #pragma unroll
            for (int r = 0; r < 8; r++) {
                int mRow = m0 + wm + i * 16 + hl * 8 + r;
                if (mRow < M) {
                    float v = acc[i][j][r] + bv;
                    if (GELU) v = geluf(v);
                    C[(size_t)mRow * ldc + nCol] = (OT)v;
                }
            }
        }
    }
}

// ---------------------------------------------------------------------------
// Elementwise / scatter / reduction kernels
// ---------------------------------------------------------------------------
__global__ void k_fill(float* __restrict__ p, float v, long n) {
    long i = (long)blockIdx.x * 256 + threadIdx.x;
    if (i < n) p[i] = v;
}

// msg = relu(h[src] + e) ; atomicAdd into agg[dst]
__global__ void k_msg_scatter(const float* __restrict__ h, const __bf16* __restrict__ e,
                              const int* __restrict__ ei, float* __restrict__ agg, int E) {
    long idx = (long)blockIdx.x * 256 + threadIdx.x;   // unit = (edge, 4 channels)
    long total = (long)E * 128;
    if (idx >= total) return;
    int eid = (int)(idx >> 7);
    int c4  = (int)(idx & 127) * 4;
    int src = ei[eid];
    int dst = ei[E + eid];
    const float*  hs = h   + (size_t)src * 512 + c4;
    const __bf16* er = e   + (size_t)eid * 512 + c4;
    float*        ar = agg + (size_t)dst * 512 + c4;
    #pragma unroll
    for (int j = 0; j < 4; j++) {
        float m = hs[j] + (float)er[j];
        m = m > 0.0f ? m : 0.0f;
        atomicAdd(ar + j, m);
    }
}

// z_bf16 = (1+eps)*h + agg   (bf16 A-operand for the following WMMA GEMM)
__global__ void k_combine(const float* __restrict__ h, const float* __restrict__ agg,
                          const float* __restrict__ epsPtr, __bf16* __restrict__ z, long n) {
    long i = (long)blockIdx.x * 256 + threadIdx.x;
    if (i < n) z[i] = (__bf16)((1.0f + epsPtr[0]) * h[i] + agg[i]);
}

// LayerNorm(z) -> gelu -> h += ; one wave32 per 512-wide row
__global__ void k_ln_gelu_res(const float* __restrict__ z,
                              const float* __restrict__ g, const float* __restrict__ b,
                              float* __restrict__ h, int M) {
    int row  = blockIdx.x * 8 + (threadIdx.x >> 5);
    int lane = threadIdx.x & 31;
    if (row >= M) return;
    const float* zr = z + (size_t)row * 512;
    float v[16], s = 0.f, s2 = 0.f;
    #pragma unroll
    for (int i = 0; i < 16; i++) {
        float t = zr[lane + i * 32];
        v[i] = t; s += t; s2 += t * t;
    }
    #pragma unroll
    for (int o = 16; o >= 1; o >>= 1) {
        s  += __shfl_xor(s,  o, 32);
        s2 += __shfl_xor(s2, o, 32);
    }
    float mu  = s * (1.0f / 512.0f);
    float var = s2 * (1.0f / 512.0f) - mu * mu;
    float inv = rsqrtf(var + 1e-5f);
    float* hr = h + (size_t)row * 512;
    #pragma unroll
    for (int i = 0; i < 16; i++) {
        int c = lane + i * 32;
        float zn = (v[i] - mu) * inv * g[c] + b[c];
        hr[c] += geluf(zn);
    }
}

// gate = gate1 . gW2 + gb2 ; one wave per node (256 elems)
__global__ void k_gate2(const float* __restrict__ g1, const float* __restrict__ gW2,
                        const float* __restrict__ gb2, float* __restrict__ gate, int M) {
    int row  = blockIdx.x * 8 + (threadIdx.x >> 5);
    int lane = threadIdx.x & 31;
    if (row >= M) return;
    const float* r = g1 + (size_t)row * 256;
    float s = 0.f;
    #pragma unroll
    for (int i = 0; i < 8; i++) { int c = lane + i * 32; s += r[c] * gW2[c]; }
    #pragma unroll
    for (int o = 16; o >= 1; o >>= 1) s += __shfl_xor(s, o, 32);
    if (lane == 0) gate[row] = s + gb2[0];
}

__global__ void k_gate_max(const float* __restrict__ gate, const int* __restrict__ batch,
                           float* __restrict__ gmax, int N) {
    int i = blockIdx.x * 256 + threadIdx.x;
    if (i < N) atomicMaxF(&gmax[batch[i]], gate[i]);
}

__global__ void k_gate_exp(const float* __restrict__ gate, const int* __restrict__ batch,
                           const float* __restrict__ gmax, float* __restrict__ ex,
                           float* __restrict__ den, int N) {
    int i = blockIdx.x * 256 + threadIdx.x;
    if (i >= N) return;
    float e = expf(gate[i] - gmax[batch[i]]);
    ex[i] = e;
    atomicAdd(&den[batch[i]], e);
}

__global__ void k_pool(const float* __restrict__ h, const int* __restrict__ batch,
                       const float* __restrict__ ex, const float* __restrict__ den,
                       float* __restrict__ g, int N) {
    long idx = (long)blockIdx.x * 256 + threadIdx.x;
    long total = (long)N * 128;
    if (idx >= total) return;
    int i  = (int)(idx >> 7);
    int c4 = (int)(idx & 127) * 4;
    int bg = batch[i];
    float alpha = ex[i] / (den[bg] + 1e-16f);
    const float* hr = h + (size_t)i  * 512 + c4;
    float*       gr = g + (size_t)bg * 512 + c4;
    #pragma unroll
    for (int j = 0; j < 4; j++) atomicAdd(gr + j, alpha * hr[j]);
}

__global__ void k_normalize(const float* __restrict__ p2, float* __restrict__ out, int B) {
    __shared__ float red[256];
    int b = blockIdx.x, t = threadIdx.x;
    const float* r = p2 + (size_t)b * 768;
    float s = 0.f;
    for (int c = t; c < 768; c += 256) { float v = r[c]; s += v * v; }
    red[t] = s;
    __syncthreads();
    for (int o = 128; o >= 1; o >>= 1) { if (t < o) red[t] += red[t + o]; __syncthreads(); }
    float inv = 1.0f / (sqrtf(red[0]) + 1e-12f);
    for (int c = t; c < 768; c += 256) out[(size_t)b * 768 + c] = r[c] * inv;
}

// ---------------------------------------------------------------------------
// Host orchestration
// ---------------------------------------------------------------------------
static inline long cdiv(long a, long b) { return (a + b - 1) / b; }

extern "C" void kernel_launch(void* const* d_in, const int* in_sizes, int n_in,
                              void* d_out, int out_size, void* d_ws, size_t ws_size,
                              hipStream_t stream) {
    const int N = in_sizes[0] / 9;      // 150000 nodes
    const int E = in_sizes[1] / 2;      // 300000 edges
    const int B = 4096;                 // graphs
    (void)n_in; (void)ws_size;

    // ---- inputs (setup_inputs dict order, params flattened in insertion order)
    const int* x     = (const int*)d_in[0];
    const int* ei    = (const int*)d_in[1];
    const int* ea    = (const int*)d_in[2];
    const int* batch = (const int*)d_in[3];
    const float* aemb[9];
    for (int i = 0; i < 9; i++) aemb[i] = (const float*)d_in[4 + i];
    const float* atom_pW = (const float*)d_in[13];
    const float* atom_pb = (const float*)d_in[14];
    const float* eemb[3];
    for (int i = 0; i < 3; i++) eemb[i] = (const float*)d_in[15 + i];
    const float* edge_pW = (const float*)d_in[18];
    const float* edge_pb = (const float*)d_in[19];
    const float *Leps[4], *LW1[4], *Lb1[4], *LW2[4], *Lb2[4], *Llng[4], *Llnb[4];
    for (int l = 0; l < 4; l++) {
        int o = 20 + l * 7;
        Leps[l] = (const float*)d_in[o + 0];
        LW1[l]  = (const float*)d_in[o + 1];
        Lb1[l]  = (const float*)d_in[o + 2];
        LW2[l]  = (const float*)d_in[o + 3];
        Lb2[l]  = (const float*)d_in[o + 4];
        Llng[l] = (const float*)d_in[o + 5];
        Llnb[l] = (const float*)d_in[o + 6];
    }
    const float* gW1 = (const float*)d_in[48];
    const float* gb1 = (const float*)d_in[49];
    const float* gW2 = (const float*)d_in[50];
    const float* gb2 = (const float*)d_in[51];
    const float* rW  = (const float*)d_in[52];
    const float* rb  = (const float*)d_in[53];
    const float* p1W = (const float*)d_in[54];
    const float* p1b = (const float*)d_in[55];
    const float* p2W = (const float*)d_in[56];
    const float* p2b = (const float*)d_in[57];

    // ---- workspace carve-out (256B aligned)
    char* cur = (char*)d_ws;
    auto alloc = [&](size_t bytes) -> void* {
        void* p = (void*)cur;
        cur += (bytes + 255) & ~(size_t)255;
        return p;
    };
    __bf16* WtAtom = (__bf16*)alloc((size_t)512 * 448 * 2);
    __bf16* WtEdge = (__bf16*)alloc((size_t)512 * 64 * 2);
    __bf16* WtL1[4]; __bf16* WtL2[4];
    for (int l = 0; l < 4; l++) {
        WtL1[l] = (__bf16*)alloc((size_t)512 * 512 * 2);
        WtL2[l] = (__bf16*)alloc((size_t)512 * 512 * 2);
    }
    __bf16* WtG1 = (__bf16*)alloc((size_t)256 * 512 * 2);
    __bf16* WtR  = (__bf16*)alloc((size_t)512 * 512 * 2);
    __bf16* WtP1 = (__bf16*)alloc((size_t)512 * 512 * 2);
    __bf16* WtP2 = (__bf16*)alloc((size_t)768 * 512 * 2);
    __bf16* Xcat = (__bf16*)alloc((size_t)N * 448 * 2);
    __bf16* Ein  = (__bf16*)alloc((size_t)E * 64 * 2);
    __bf16* Ebuf = (__bf16*)alloc((size_t)E * 512 * 2);
    float*  h    = (float*)alloc((size_t)N * 512 * 4);
    float*  agg  = (float*)alloc((size_t)N * 512 * 4);
    __bf16* zb   = (__bf16*)alloc((size_t)N * 512 * 2);   // GEMM-A operand, bf16
    __bf16* t1b  = (__bf16*)alloc((size_t)N * 512 * 2);   // GEMM-A operand, bf16
    float*  z2   = (float*)alloc((size_t)N * 512 * 4);    // LayerNorm input, f32
    float*  g1b  = (float*)alloc((size_t)N * 256 * 4);
    float*  gate = (float*)alloc((size_t)N * 4);
    float*  ex   = (float*)alloc((size_t)N * 4);
    float*  gmax = (float*)alloc((size_t)B * 4);
    float*  den  = (float*)alloc((size_t)B * 4);
    float*  gbuf = (float*)alloc((size_t)B * 512 * 4);
    float*  rbuf = (float*)alloc((size_t)B * 512 * 4);
    float*  p1bf = (float*)alloc((size_t)B * 512 * 4);
    float*  p2bf = (float*)alloc((size_t)B * 768 * 4);

    // ---- weight convert/transpose to bf16 [N][Kp]
    auto wcv = [&](const float* W, __bf16* Wt, int K, int Nn, int Kp) {
        long total = (long)Nn * Kp;
        k_wconv<<<dim3((unsigned)cdiv(total, 256)), dim3(256), 0, stream>>>(W, Wt, K, Nn, Kp);
    };
    wcv(atom_pW, WtAtom, 432, 512, 448);
    wcv(edge_pW, WtEdge, 48, 512, 64);
    for (int l = 0; l < 4; l++) { wcv(LW1[l], WtL1[l], 512, 512, 512);
                                  wcv(LW2[l], WtL2[l], 512, 512, 512); }
    wcv(gW1, WtG1, 512, 256, 512);
    wcv(rW,  WtR,  512, 512, 512);
    wcv(p1W, WtP1, 512, 512, 512);
    wcv(p2W, WtP2, 512, 768, 512);

    // ---- encoders
    k_embed_atoms<<<dim3((unsigned)cdiv((long)N * 448, 256)), dim3(256), 0, stream>>>(
        x, aemb[0], aemb[1], aemb[2], aemb[3], aemb[4], aemb[5], aemb[6], aemb[7], aemb[8],
        Xcat, N);
    k_embed_edges<<<dim3((unsigned)cdiv((long)E * 64, 256)), dim3(256), 0, stream>>>(
        ea, eemb[0], eemb[1], eemb[2], Ein, E);

    // h = gelu(Xcat @ atom_pW + b)   [N,512], K=448 (padded)
    k_gemm<true, __bf16, float><<<dim3(2, (unsigned)cdiv(N, 128)), dim3(256), 0, stream>>>(
        Xcat, 448, WtAtom, atom_pb, h, 512, N, 512, 448);
    // e = gelu(Ein @ edge_pW + b)    [E,512] bf16, K=64 (padded)
    k_gemm<true, __bf16, __bf16><<<dim3(2, (unsigned)cdiv(E, 128)), dim3(256), 0, stream>>>(
        Ein, 64, WtEdge, edge_pb, Ebuf, 512, E, 512, 64);

    // ---- 4 GINE layers
    for (int l = 0; l < 4; l++) {
        k_fill<<<dim3((unsigned)cdiv((long)N * 512, 256)), dim3(256), 0, stream>>>(
            agg, 0.0f, (long)N * 512);
        k_msg_scatter<<<dim3((unsigned)cdiv((long)E * 128, 256)), dim3(256), 0, stream>>>(
            h, Ebuf, ei, agg, E);
        k_combine<<<dim3((unsigned)cdiv((long)N * 512, 256)), dim3(256), 0, stream>>>(
            h, agg, Leps[l], zb, (long)N * 512);
        k_gemm<true, __bf16, __bf16><<<dim3(2, (unsigned)cdiv(N, 128)), dim3(256), 0, stream>>>(
            zb, 512, WtL1[l], Lb1[l], t1b, 512, N, 512, 512);
        k_gemm<false, __bf16, float><<<dim3(2, (unsigned)cdiv(N, 128)), dim3(256), 0, stream>>>(
            t1b, 512, WtL2[l], Lb2[l], z2, 512, N, 512, 512);
        k_ln_gelu_res<<<dim3((unsigned)cdiv(N, 8)), dim3(256), 0, stream>>>(
            z2, Llng[l], Llnb[l], h, N);
    }

    // ---- attentional aggregation
    k_gemm<true, float, float><<<dim3(1, (unsigned)cdiv(N, 128)), dim3(256), 0, stream>>>(
        h, 512, WtG1, gb1, g1b, 256, N, 256, 512);
    k_gate2<<<dim3((unsigned)cdiv(N, 8)), dim3(256), 0, stream>>>(g1b, gW2, gb2, gate, N);
    k_fill<<<dim3((unsigned)cdiv((long)B, 256)), dim3(256), 0, stream>>>(gmax, -INFINITY, B);
    k_fill<<<dim3((unsigned)cdiv((long)B, 256)), dim3(256), 0, stream>>>(den, 0.0f, B);
    k_gate_max<<<dim3((unsigned)cdiv(N, 256)), dim3(256), 0, stream>>>(gate, batch, gmax, N);
    k_gate_exp<<<dim3((unsigned)cdiv(N, 256)), dim3(256), 0, stream>>>(gate, batch, gmax, ex, den, N);
    k_fill<<<dim3((unsigned)cdiv((long)B * 512, 256)), dim3(256), 0, stream>>>(
        gbuf, 0.0f, (long)B * 512);
    k_pool<<<dim3((unsigned)cdiv((long)N * 128, 256)), dim3(256), 0, stream>>>(
        h, batch, ex, den, gbuf, N);

    // ---- graph head
    k_gemm<true, float, float><<<dim3(2, (unsigned)cdiv(B, 128)), dim3(256), 0, stream>>>(
        gbuf, 512, WtR, rb, rbuf, 512, B, 512, 512);
    k_gemm<true, float, float><<<dim3(2, (unsigned)cdiv(B, 128)), dim3(256), 0, stream>>>(
        rbuf, 512, WtP1, p1b, p1bf, 512, B, 512, 512);
    k_gemm<false, float, float><<<dim3(3, (unsigned)cdiv(B, 128)), dim3(256), 0, stream>>>(
        p1bf, 512, WtP2, p2b, p2bf, 768, B, 768, 512);
    k_normalize<<<dim3(B), dim3(256), 0, stream>>>(p2bf, (float*)d_out, B);
    (void)out_size;
}
